// U2NET_24996709663122
// MI455X (gfx1250) — compile-verified, matrix operand order checked by reading, not run on hardware
//
#include <hip/hip_runtime.h>

// =====================================================================
// Sparse U2Net (MinkUNet stack) forward for MI455X / gfx1250.
// Core compute: gather-GEMM via V_WMMA_F32_16X16X32_BF16 (f32 accum),
// with the gathered A tile staged once per block in LDS (bf16) and
// 2x M-blocking per wave so each B fragment feeds two WMMAs.
// Input flattening assumed: jax tree_flatten order (sorted dict keys).
// =====================================================================

typedef __attribute__((ext_vector_type(16))) __bf16 v16bf;
typedef __attribute__((ext_vector_type(8)))  __bf16 v8bf;
typedef __attribute__((ext_vector_type(8)))  float  v8f;
typedef __attribute__((ext_vector_type(4)))  float  v4f;

#define AST 40   // LDS A-tile row stride in bf16 (80B: 16B-aligned, bank-spread)

// ---------------------------------------------------------------------
// Weight transpose+convert: W[K,Ci,Co] f32 -> Wt[K,Co,Ci] bf16
// ---------------------------------------------------------------------
__global__ void k_wtrans(const float* __restrict__ W, __bf16* __restrict__ Wt,
                         int K, int Ci, int Co) {
  int i = blockIdx.x * blockDim.x + threadIdx.x;
  int tot = K * Ci * Co;
  if (i >= tot) return;
  int co = i % Co;
  int t  = i / Co;
  int ci = t % Ci;
  int k  = t / Ci;
  Wt[((long)k * Co + co) * Ci + ci] = (__bf16)W[i];
}

// ---------------------------------------------------------------------
// Gather-GEMM: out[Nout,Co] = sum_k gather(x, nbr[:,k]) @ W[k]
//   nbr == nullptr  -> dense (idx = row); bias != nullptr -> added at store
// Block = 128 threads (4 waves). Block tile: M=32 rows x N=64 cols.
// Per (k, cin-chunk): all 128 threads gather/convert the 32x32 A tile
// into LDS (natural cin order, bf16); each wave then reads two A-frags
// (rows lane%16 and lane%16+16) and one B-frag, issuing two WMMAs into
// independent accumulators.
// Frag layouts (ISA 7.12.2):
//  A: lane L holds row M=L%16; elem j -> cin = c0 + (j<8 ? j : j+8) + 8*(L/16)
//  B: lane L holds col N=L%16; same k-packing from Wt[k][col][*]
//  C: VGPR r -> row r + 8*(L/16), col L%16
// ---------------------------------------------------------------------
__global__ void __launch_bounds__(128)
k_gconv(const float* __restrict__ x, const __bf16* __restrict__ Wt,
        const int* __restrict__ nbr, const float* __restrict__ bias,
        float* __restrict__ out, int N, int K, int Ci, int Co) {
  __shared__ __bf16 As[32 * AST];

  const int lane = threadIdx.x;                     // 0..31
  const int tid  = threadIdx.y * 32 + lane;         // 0..127
  const int m0   = blockIdx.x * 32;
  const int n0   = (blockIdx.y * 4 + threadIdx.y) * 16;
  const int mn   = lane & 15;
  const int h    = lane >> 4;
  const int col  = n0 + mn;

  // cooperative-fill assignment: row = tid/4 (0..31), quarter q = tid%4
  const int frow = tid >> 2;
  const int fq   = tid & 3;
  const int grow = m0 + frow;

  v8f acc0 = {};
  v8f acc1 = {};

  for (int k = 0; k < K; ++k) {
    const __bf16* wr = (col < Co) ? (Wt + ((long)k * Co + col) * Ci) : nullptr;
    const int gidx = (grow < N) ? (nbr ? nbr[(long)grow * K + k] : grow) : -1;

    for (int c0 = 0; c0 < Ci; c0 += 32) {
      // ---- stage gathered 32x32 A tile to LDS as bf16 ----
      {
        v8bf v = {};
        if (gidx >= 0) {
          if (c0 + 32 <= Ci) {
            const float* xr = x + (long)gidx * Ci + c0 + fq * 8;
            const v4f f0 = *(const v4f*)(xr);
            const v4f f1 = *(const v4f*)(xr + 4);
            v[0] = (__bf16)f0.x; v[1] = (__bf16)f0.y;
            v[2] = (__bf16)f0.z; v[3] = (__bf16)f0.w;
            v[4] = (__bf16)f1.x; v[5] = (__bf16)f1.y;
            v[6] = (__bf16)f1.z; v[7] = (__bf16)f1.w;
          } else {
#pragma unroll
            for (int j = 0; j < 8; ++j) {
              const int ci = c0 + fq * 8 + j;
              if (ci < Ci) v[j] = (__bf16)x[(long)gidx * Ci + ci];
            }
          }
        }
        *(v8bf*)(As + frow * AST + fq * 8) = v;
      }
      __syncthreads();

      // ---- B fragment from transposed bf16 weights ----
      v16bf b = {};
      if (wr) {
        if (c0 + 32 <= Ci) {
          const v8bf b0 = *(const v8bf*)(wr + c0 + 8 * h);
          const v8bf b1 = *(const v8bf*)(wr + c0 + 16 + 8 * h);
#pragma unroll
          for (int j = 0; j < 8; ++j) { b[j] = b0[j]; b[8 + j] = b1[j]; }
        } else {
#pragma unroll
          for (int j = 0; j < 16; ++j) {
            const int ci = c0 + ((j < 8) ? j : (j + 8)) + 8 * h;
            if (ci < Ci) b[j] = wr[ci];
          }
        }
      }

      // ---- two A fragments from LDS (rows mn and mn+16) ----
      const __bf16* r0p = As + mn * AST;
      const __bf16* r1p = As + (mn + 16) * AST;
      const v8bf a0lo = *(const v8bf*)(r0p + 8 * h);
      const v8bf a0hi = *(const v8bf*)(r0p + 16 + 8 * h);
      const v8bf a1lo = *(const v8bf*)(r1p + 8 * h);
      const v8bf a1hi = *(const v8bf*)(r1p + 16 + 8 * h);
      v16bf a0, a1;
#pragma unroll
      for (int j = 0; j < 8; ++j) {
        a0[j] = a0lo[j]; a0[8 + j] = a0hi[j];
        a1[j] = a1lo[j]; a1[8 + j] = a1hi[j];
      }

      acc0 = __builtin_amdgcn_wmma_f32_16x16x32_bf16(
          false, a0, false, b, (short)0, acc0, false, false);
      acc1 = __builtin_amdgcn_wmma_f32_16x16x32_bf16(
          false, a1, false, b, (short)0, acc1, false, false);

      __syncthreads();
    }
  }

  if (col < Co) {
    const float bv = bias ? bias[col] : 0.0f;
#pragma unroll
    for (int r = 0; r < 8; ++r) {
      const int mr0 = m0 + r + 8 * h;
      const int mr1 = mr0 + 16;
      if (mr0 < N) out[(long)mr0 * Co + col] = acc0[r] + bv;
      if (mr1 < N) out[(long)mr1 * Co + col] = acc1[r] + bv;
    }
  }
}

// ---------------------------------------------------------------------
// BatchNorm (training-mode batch stats), 3 kernels
// ---------------------------------------------------------------------
__global__ void __launch_bounds__(256)
k_bnstat1(const float* __restrict__ x, float* __restrict__ part, int N, int C) {
  __shared__ float s[256];
  __shared__ float s2[256];
  const int t = threadIdx.x;
  s[t] = 0.f; s2[t] = 0.f;
  __syncthreads();
  const int nb = gridDim.x;
  long rb = ((long)N + nb - 1) / nb;
  long r0 = (long)blockIdx.x * rb;
  long r1 = r0 + rb; if (r1 > N) r1 = N;
  if (r0 > r1) r0 = r1;
  const long begin = r0 * C, end = r1 * C;
  for (long i = begin + t; i < end; i += blockDim.x) {
    const float v = x[i];
    const int ch = (int)(i % C);
    atomicAdd(&s[ch], v);        // ds_add_f32
    atomicAdd(&s2[ch], v * v);
  }
  __syncthreads();
  if (t < C) {
    part[(long)blockIdx.x * 512 + t]       = s[t];
    part[(long)blockIdx.x * 512 + 256 + t] = s2[t];
  }
}

__global__ void __launch_bounds__(256)
k_bnstat2(const float* __restrict__ part, const float* __restrict__ g,
          const float* __restrict__ b, float* __restrict__ ss,
          int C, int nb, int N) {
  const int ch = threadIdx.x;
  if (ch >= C) return;
  float s = 0.f, s2 = 0.f;
  for (int i = 0; i < nb; ++i) {
    s  += part[(long)i * 512 + ch];
    s2 += part[(long)i * 512 + 256 + ch];
  }
  const float inv = 1.0f / (float)N;
  const float mu  = s * inv;
  const float var = s2 * inv - mu * mu;
  const float sc  = g[ch] * rsqrtf(var + 1e-5f);
  ss[ch]       = sc;
  ss[256 + ch] = b[ch] - mu * sc;
}

__global__ void __launch_bounds__(256)
k_bnapply(float* __restrict__ x, const float* __restrict__ ss,
          long tot, int C, int relu) {
  const long i = (long)blockIdx.x * blockDim.x + threadIdx.x;
  if (i >= tot) return;
  const int ch = (int)(i % C);
  float v = x[i] * ss[ch] + ss[256 + ch];
  if (relu) v = fmaxf(v, 0.0f);
  x[i] = v;
}

// ---------------------------------------------------------------------
// Misc elementwise kernels
// ---------------------------------------------------------------------
__global__ void __launch_bounds__(256)
k_addrelu(float* __restrict__ h, const float* __restrict__ s, long tot) {
  const long i = (long)blockIdx.x * blockDim.x + threadIdx.x;
  if (i >= tot) return;
  h[i] = fmaxf(h[i] + s[i], 0.0f);
}

__global__ void __launch_bounds__(256)
k_copycols(const float* __restrict__ in, float* __restrict__ out,
           long tot, int Ci, int ldo, int off) {
  const long i = (long)blockIdx.x * blockDim.x + threadIdx.x;
  if (i >= tot) return;
  const long n = i / Ci;
  const int  c = (int)(i % Ci);
  out[n * ldo + off + c] = in[i];
}

// up_conv -> gconv lowering: un[i,k] = (off[i]==k) ? par[i] : -1
__global__ void __launch_bounds__(256)
k_upnbr(const int* __restrict__ par, const int* __restrict__ ofs,
        int* __restrict__ un, int N) {
  const int i = blockIdx.x * blockDim.x + threadIdx.x;
  if (i >= N) return;
  const int p = par[i], o = ofs[i];
#pragma unroll
  for (int k = 0; k < 8; ++k) un[(long)i * 8 + k] = (o == k) ? p : -1;
}

// =====================================================================
// Host-side orchestration
// =====================================================================

struct BNp   { const float* g; const float* b; int c; };
struct Convp { const float* w; int K, ci, co; };
struct Basicp { BNp bn; Convp c; };
struct Resp  { BNp bn1, bn2, bnd; Convp c1, c2; const float* cd; int ci, co; bool hasd; };
struct MinkDec { Resp r1, r2; Basicp up; };
struct MinkEnc { Basicp down; Resp r1, r2; };
struct Minkp { BNp bns1, bns2; MinkDec dec[2]; MinkEnc enc[2]; Convp stem1, stem2; int nel; };

struct Cursor {
  void* const* d; const int* sz; int i;
  const float* f()  { return (const float*)d[i++]; }
  const int*   ii() { return (const int*)d[i++]; }
};

static BNp P_bn(Cursor& c, int ch) { BNp r; r.b = c.f(); r.g = c.f(); r.c = ch; return r; } // sorted: b,g
static Convp P_conv(Cursor& c, int K, int ci, int co) { Convp r; r.w = c.f(); r.K = K; r.ci = ci; r.co = co; return r; }
static Basicp P_basic(Cursor& c, int K, int ci, int co) { Basicp r; r.bn = P_bn(c, co); r.c = P_conv(c, K, ci, co); return r; } // sorted: bn,c
static Resp P_res(Cursor& c, int ci, int co) {
  Resp r; r.ci = ci; r.co = co; r.hasd = (ci != co); r.cd = nullptr;
  r.bn1 = P_bn(c, co); r.bn2 = P_bn(c, co);
  if (r.hasd) r.bnd = P_bn(c, co);              // sorted: bn1,bn2,bnd,c1,c2,cd
  r.c1 = P_conv(c, 27, ci, co); r.c2 = P_conv(c, 27, co, co);
  if (r.hasd) r.cd = c.f();
  return r;
}
static Minkp P_mink(Cursor& c, int inc, int nel, const int* cs) {
  Minkp p; p.nel = nel;                         // sorted: bns1,bns2,dec,enc,stem1,stem2
  p.bns1 = P_bn(c, cs[0]); p.bns2 = P_bn(c, cs[0]);
  for (int i = 0; i < nel; ++i) {               // dec entry sorted: r1,r2,up
    const int co = cs[nel + i + 1];
    p.dec[i].r1 = P_res(c, co + cs[nel - i - 1], co);
    p.dec[i].r2 = P_res(c, co, co);
    p.dec[i].up = P_basic(c, 8, cs[nel + i], co);
  }
  for (int i = 0; i < nel; ++i) {               // enc entry sorted: down,r1,r2
    p.enc[i].down = P_basic(c, 8, cs[i], cs[i]);
    p.enc[i].r1   = P_res(c, cs[i], cs[i + 1]);
    p.enc[i].r2   = P_res(c, cs[i + 1], cs[i + 1]);
  }
  p.stem1 = P_conv(c, 27, inc, cs[0]);
  p.stem2 = P_conv(c, 27, cs[0], cs[0]);
  return p;
}

struct Maps { const int* nbr[3]; const int* down[2]; const int* up_par[2]; const int* up_off[2]; };

struct Ctx {
  hipStream_t st;
  char* ws;
  __bf16* wbuf; float* bnpart; float* bnss; int* upnbr;
  size_t abase, aend, acap;
  struct Blk { size_t off, sz; int used; } blk[512];
  int nblk;

  float* alloc(size_t elems) {
    size_t bytes = (elems * 4 + 255) & ~(size_t)255;
    int best = -1;
    for (int i = 0; i < nblk; ++i)
      if (!blk[i].used && blk[i].sz >= bytes && (best < 0 || blk[i].sz < blk[best].sz)) best = i;
    if (best >= 0) { blk[best].used = 1; return (float*)(ws + abase + blk[best].off); }
    size_t off = aend; aend += bytes;
    if (nblk < 512) { blk[nblk].off = off; blk[nblk].sz = bytes; blk[nblk].used = 1; ++nblk; }
    return (float*)(ws + abase + off);
  }
  void rel(const float* p) {
    size_t off = (size_t)((const char*)p - (ws + abase));
    for (int i = 0; i < nblk; ++i) if (blk[i].off == off && blk[i].used) { blk[i].used = 0; break; }
    while (nblk > 0 && !blk[nblk - 1].used) { aend = blk[nblk - 1].off; --nblk; }
  }
};

static void run_gconv(Ctx& C, const float* x, const int* nbr, int Nout,
                      const Convp& cv, float* out, const float* bias = nullptr) {
  const int tot = cv.K * cv.ci * cv.co;
  k_wtrans<<<(tot + 255) / 256, 256, 0, C.st>>>(cv.w, C.wbuf, cv.K, cv.ci, cv.co);
  dim3 g((Nout + 31) / 32, (cv.co + 63) / 64), b(32, 4, 1);
  k_gconv<<<g, b, 0, C.st>>>(x, C.wbuf, nbr, bias, out, Nout, cv.K, cv.ci, cv.co);
}

static void run_bn(Ctx& C, float* x, int N, const BNp& bn, bool relu) {
  const int NB = 64;
  k_bnstat1<<<NB, 256, 0, C.st>>>(x, C.bnpart, N, bn.c);
  k_bnstat2<<<1, 256, 0, C.st>>>(C.bnpart, bn.g, bn.b, C.bnss, bn.c, NB, N);
  const long tot = (long)N * bn.c;
  k_bnapply<<<(unsigned)((tot + 255) / 256), 256, 0, C.st>>>(x, C.bnss, tot, bn.c, relu ? 1 : 0);
}

static void run_addrelu(Ctx& C, float* h, const float* s, long tot) {
  k_addrelu<<<(unsigned)((tot + 255) / 256), 256, 0, C.st>>>(h, s, tot);
}

static void run_copycols(Ctx& C, const float* in, float* out, int N, int Ci, int ldo, int off) {
  const long tot = (long)N * Ci;
  k_copycols<<<(unsigned)((tot + 255) / 256), 256, 0, C.st>>>(in, out, tot, Ci, ldo, off);
}

static float* res_block(Ctx& C, const float* x, int N, const int* nbr, const Resp& r) {
  float* h1 = C.alloc((size_t)N * r.co);
  run_gconv(C, x, nbr, N, r.c1, h1);
  run_bn(C, h1, N, r.bn1, true);
  float* h = C.alloc((size_t)N * r.co);
  run_gconv(C, h1, nbr, N, r.c2, h);
  run_bn(C, h, N, r.bn2, false);
  C.rel(h1);
  if (r.hasd) {
    float* s = C.alloc((size_t)N * r.co);
    Convp cd; cd.w = r.cd; cd.K = 1; cd.ci = r.ci; cd.co = r.co;
    run_gconv(C, x, nullptr, N, cd, s);
    run_bn(C, s, N, r.bnd, false);
    run_addrelu(C, h, s, (long)N * r.co);
    C.rel(s);
  } else {
    run_addrelu(C, h, x, (long)N * r.co);
  }
  return h;
}

static float* up_conv(Ctx& C, const float* x, int l, const Basicp& up,
                      const Maps& M, const int* Nlev) {
  const int Nout = Nlev[l];
  k_upnbr<<<(Nout + 255) / 256, 256, 0, C.st>>>(M.up_par[l], M.up_off[l], C.upnbr, Nout);
  float* o = C.alloc((size_t)Nout * up.c.co);
  run_gconv(C, x, C.upnbr, Nout, up.c, o);
  run_bn(C, o, Nout, up.bn, true);
  return o;
}

static float* minkunet(Ctx& C, const float* x, const Minkp& p, int lvl, int nel,
                       const Maps& M, const int* Nlev) {
  const int N = Nlev[lvl];
  float* h1 = C.alloc((size_t)N * p.stem1.co);
  run_gconv(C, x, M.nbr[lvl], N, p.stem1, h1);
  run_bn(C, h1, N, p.bns1, true);
  float* h2 = C.alloc((size_t)N * p.stem2.co);
  run_gconv(C, h1, M.nbr[lvl], N, p.stem2, h2);
  run_bn(C, h2, N, p.bns2, true);
  C.rel(h1);

  float* xs[3]; int chs[3];
  xs[0] = h2; chs[0] = p.stem2.co;
  for (int i = 0; i < nel; ++i) {
    const int Nd = Nlev[lvl + i + 1];
    float* d = C.alloc((size_t)Nd * p.enc[i].down.c.co);
    run_gconv(C, xs[i], M.down[lvl + i], Nd, p.enc[i].down.c, d);
    run_bn(C, d, Nd, p.enc[i].down.bn, true);
    float* r1 = res_block(C, d, Nd, M.nbr[lvl + i + 1], p.enc[i].r1); C.rel(d);
    float* r2 = res_block(C, r1, Nd, M.nbr[lvl + i + 1], p.enc[i].r2); C.rel(r1);
    xs[i + 1] = r2; chs[i + 1] = p.enc[i].r2.co;
  }

  float* y = xs[nel];
  for (int i = 0; i < nel; ++i) {
    const int l  = lvl + nel - i - 1;
    const int Nl = Nlev[l];
    float* u = up_conv(C, y, l, p.dec[i].up, M, Nlev);
    C.rel(y);
    const int Cu = p.dec[i].up.c.co;
    const int Cs = chs[nel - i - 1];
    const int Ct = Cu + Cs;
    float* cat = C.alloc((size_t)Nl * Ct);
    run_copycols(C, u, cat, Nl, Cu, Ct, 0);
    run_copycols(C, xs[nel - i - 1], cat, Nl, Cs, Ct, Cu);
    C.rel(u); C.rel(xs[nel - i - 1]);
    float* r1 = res_block(C, cat, Nl, M.nbr[l], p.dec[i].r1); C.rel(cat);
    y = res_block(C, r1, Nl, M.nbr[l], p.dec[i].r2); C.rel(r1);
  }
  return y;
}

extern "C" void kernel_launch(void* const* d_in, const int* in_sizes, int n_in,
                              void* d_out, int out_size, void* d_ws, size_t ws_size,
                              hipStream_t stream) {
  (void)n_in; (void)out_size;
  Cursor c; c.d = d_in; c.sz = in_sizes; c.i = 0;

  // ---- flatten order: feats; maps{down,nbr,up_off,up_par}; params sorted ----
  const float* feats = c.f();
  Maps M;
  M.down[0] = c.ii(); M.down[1] = c.ii();
  int Nlev[3];
  for (int l = 0; l < 3; ++l) { Nlev[l] = in_sizes[c.i] / 27; M.nbr[l] = c.ii(); }
  M.up_off[0] = c.ii(); M.up_off[1] = c.ii();
  M.up_par[0] = c.ii(); M.up_par[1] = c.ii();

  const float* cls_b = c.f();
  const float* cls_w = c.f();
  static const int CSv[5] = {32, 64, 128, 128, 96};
  static const int CS1[3] = {64, 128, 128};
  Minkp dec0 = P_mink(c, 256, 1, CS1);
  Minkp dec1 = P_mink(c, 224, 2, CSv);
  Minkp enc0 = P_mink(c, 4, 2, CSv);
  Minkp enc1 = P_mink(c, 96, 1, CS1);
  Basicp idec0 = P_basic(c, 8, 128, 128);
  Basicp ienc0 = P_basic(c, 8, 96, 96);
  Resp   resp  = P_res(c, 128, 128);

  // ---- workspace layout ----
  Ctx C; C.st = stream; C.ws = (char*)d_ws; C.nblk = 0; C.aend = 0;
  size_t off = 0;
  C.wbuf   = (__bf16*)(C.ws + off); off += (size_t)2 << 20;          // max conv = 27*192*128 bf16
  C.bnpart = (float*)(C.ws + off);  off += (size_t)64 * 512 * 4;
  C.bnss   = (float*)(C.ws + off);  off += (size_t)512 * 4;
  C.upnbr  = (int*)(C.ws + off);    off += (size_t)Nlev[0] * 8 * 4;
  off = (off + 255) & ~(size_t)255;
  C.abase = off; C.acap = (ws_size > off) ? (ws_size - off) : 0;

  // ---- u2net forward ----
  float* xs0 = minkunet(C, feats, enc0, 0, 2, M, Nlev);              // [N0,96]

  float* h = C.alloc((size_t)Nlev[1] * 96);
  run_gconv(C, xs0, M.down[0], Nlev[1], ienc0.c, h);
  run_bn(C, h, Nlev[1], ienc0.bn, true);                             // [N1,96]

  float* xs1 = minkunet(C, h, enc1, 1, 1, M, Nlev); C.rel(h);        // [N1,128]

  float* y = res_block(C, xs1, Nlev[1], M.nbr[1], resp);             // [N1,128]
  float* cat = C.alloc((size_t)Nlev[1] * 256);
  run_copycols(C, y, cat, Nlev[1], 128, 256, 0);
  run_copycols(C, xs1, cat, Nlev[1], 128, 256, 128);
  C.rel(y); C.rel(xs1);

  float* y2 = minkunet(C, cat, dec0, 1, 1, M, Nlev); C.rel(cat);     // [N1,128]

  float* ys0 = up_conv(C, y2, 0, idec0, M, Nlev); C.rel(y2);         // [N0,128]

  float* cat2 = C.alloc((size_t)Nlev[0] * 224);
  run_copycols(C, ys0, cat2, Nlev[0], 128, 224, 0);
  run_copycols(C, xs0, cat2, Nlev[0], 96, 224, 128);
  C.rel(xs0);

  float* ys1 = minkunet(C, cat2, dec1, 0, 2, M, Nlev); C.rel(cat2);  // [N0,96]

  float* z = C.alloc((size_t)Nlev[0] * 224);
  run_copycols(C, ys0, z, Nlev[0], 128, 224, 0);
  run_copycols(C, ys1, z, Nlev[0], 96, 224, 128);
  C.rel(ys0); C.rel(ys1);

  Convp cls; cls.w = cls_w; cls.K = 1; cls.ci = 224; cls.co = 20;
  run_gconv(C, z, nullptr, Nlev[0], cls, (float*)d_out, cls_b);      // [N0,20]
  C.rel(z);
}
